// PointConvCentroid_42073499631903
// MI455X (gfx1250) — matrix operand order repaired and checked by reading, not compiled, not run on hardware
//
#include <hip/hip_runtime.h>
#include <hip/hip_bf16.h>

typedef __attribute__((ext_vector_type(16))) _Float16 v16h;
typedef __attribute__((ext_vector_type(8)))  _Float16 v8h;
typedef __attribute__((ext_vector_type(8)))  float    v8f;

#define BDIM   16
#define N1DIM  512
#define N2DIM  8192
#define CDIM   128
#define KNN    32
#define C0PAD  160   // 131 padded up to multiple of 32
#define C1     128
#define C2     256
#define FDIM   2048  // 8 * C2
#define LEAK   0.1f

#define WMMA_F16(a, b, c) \
  __builtin_amdgcn_wmma_f32_16x16x32_f16(false, (a), false, (b), (short)0, (c), false, false)

// A-fragment (16x32 f16, M x K), memory row-major [M][K] with row stride ld.
// ISA layout: lane l -> row M=l&15; elements K = {kb..kb+7, kb+16..kb+23}, kb=(l>=16)?8:0.
static __device__ __forceinline__ v16h frag_a(const _Float16* base, int ld) {
  int l   = threadIdx.x & 31;
  int row = l & 15;
  int kb  = (l & 16) ? 8 : 0;
  const _Float16* q = base + row * ld + kb;
  v8h lo = *(const v8h*)(q);
  v8h hi = *(const v8h*)(q + 16);
  v16h r;
#pragma unroll
  for (int j = 0; j < 8; ++j) { r[j] = lo[j]; r[8 + j] = hi[j]; }
  return r;
}

// B-fragment (32x16 f16, K x N), memory stored as [N][K] (weight layout) with row stride ld.
// ISA layout: lane l -> col N=l&15; elements K = {kb..kb+7, kb+16..kb+23}, kb=(l>=16)?8:0.
static __device__ __forceinline__ v16h frag_b(const _Float16* base, int ld) {
  int l   = threadIdx.x & 31;
  int col = l & 15;
  int kb  = (l & 16) ? 8 : 0;
  const _Float16* q = base + col * ld + kb;
  v8h lo = *(const v8h*)(q);
  v8h hi = *(const v8h*)(q + 16);
  v16h r;
#pragma unroll
  for (int j = 0; j < 8; ++j) { r[j] = lo[j]; r[8 + j] = hi[j]; }
  return r;
}

// A-fragment where logical element (m,k) lives at base[k*ld + m] (transposed source).
static __device__ __forceinline__ v16h frag_a_t(const _Float16* base, int ld) {
  int l   = threadIdx.x & 31;
  int row = l & 15;
  int kb  = (l & 16) ? 8 : 0;
  v16h r;
#pragma unroll
  for (int j = 0; j < 8; ++j) {
    r[j]     = base[(kb + j) * ld + row];
    r[8 + j] = base[(kb + 16 + j) * ld + row];
  }
  return r;
}

static __device__ __forceinline__ float lrelu(float v) {
  return (v > 0.f) ? v : LEAK * v;
}

// ---------------------------------------------------------------------------
// Kernel 0: convert weights to f16 (w_mlp0 zero-padded 131 -> 160 cols)
// ---------------------------------------------------------------------------
__global__ void __launch_bounds__(256)
cvt_weights_kernel(const float* w0, const float* w1, const float* w2, const float* wl,
                   _Float16* w0h, _Float16* w1h, _Float16* w2h, _Float16* wlh) {
  const int s0 = C1 * C0PAD;   // 20480
  const int s1 = C1 * C1;      // 16384
  const int s2 = C2 * C1;      // 32768
  const int s3 = C2 * FDIM;    // 524288
  int i = blockIdx.x * 256 + threadIdx.x;
  if (i < s0) {
    int o = i / C0PAD, c = i % C0PAD;
    w0h[i] = (c < 131) ? (_Float16)w0[o * 131 + c] : (_Float16)0.f;
  } else if (i < s0 + s1) {
    int j = i - s0;           w1h[j] = (_Float16)w1[j];
  } else if (i < s0 + s1 + s2) {
    int j = i - s0 - s1;      w2h[j] = (_Float16)w2[j];
  } else if (i < s0 + s1 + s2 + s3) {
    int j = i - s0 - s1 - s2; wlh[j] = (_Float16)wl[j];
  }
}

// ---------------------------------------------------------------------------
// Kernel 1: KNN (32 smallest squared distances, index tie-break ascending)
// One 256-thread block per (b, n1); distances cached in LDS.
// ---------------------------------------------------------------------------
__global__ void __launch_bounds__(256)
knn_kernel(const float* xyz1, const float* xyz2, int* knn_out) {
  __shared__ float dist[N2DIM];
  __shared__ float rv[256];
  __shared__ int   ri[256];
  int p  = blockIdx.x;
  int b  = p >> 9;
  int n1 = p & (N1DIM - 1);

  float x0 = xyz1[(size_t)b * 3 * N1DIM + 0 * N1DIM + n1];
  float y0 = xyz1[(size_t)b * 3 * N1DIM + 1 * N1DIM + n1];
  float z0 = xyz1[(size_t)b * 3 * N1DIM + 2 * N1DIM + n1];
  float q1 = x0 * x0 + y0 * y0 + z0 * z0;
  const float* xs = xyz2 + (size_t)b * 3 * N2DIM;

  for (int j = threadIdx.x; j < N2DIM; j += 256) {
    float x = xs[j], y = xs[N2DIM + j], z = xs[2 * N2DIM + j];
    dist[j] = q1 + (x * x + y * y + z * z) - 2.f * (x0 * x + y0 * y + z0 * z);
  }
  __syncthreads();

  for (int k = 0; k < KNN; ++k) {
    float best = 3.4e38f; int bi = N2DIM;
    for (int j = threadIdx.x; j < N2DIM; j += 256) {
      float d = dist[j];
      if (d < best) { best = d; bi = j; }   // ascending scan: lowest index wins ties
    }
    rv[threadIdx.x] = best; ri[threadIdx.x] = bi;
    __syncthreads();
    for (int s = 128; s > 0; s >>= 1) {
      if (threadIdx.x < s) {
        float ov = rv[threadIdx.x + s]; int oi = ri[threadIdx.x + s];
        float mv = rv[threadIdx.x];     int mi = ri[threadIdx.x];
        if (ov < mv || (ov == mv && oi < mi)) { rv[threadIdx.x] = ov; ri[threadIdx.x] = oi; }
      }
      __syncthreads();
    }
    if (threadIdx.x == 0) {
      int win = ri[0];
      knn_out[(size_t)p * KNN + k] = win;
      dist[win] = 3.4e38f;
    }
    __syncthreads();
  }
}

// ---------------------------------------------------------------------------
// Kernel 2: gather + MLP0/1/2 (WMMA, 2x2 register blocked) + WeightNet +
//           agg (WMMA) -> flat[P][2048].  One 128-thread block per point.
// ---------------------------------------------------------------------------
static __device__ __forceinline__ void
mlp_layer(const _Float16* in, int ldk, int ksteps, const _Float16* w,
          _Float16* out, int nout, int wave, int lane) {
  int nblk = nout >> 5;                  // # of 2-wide N-tile groups
  int ncol = lane & 15;
  int mrow = (lane & 16) ? 8 : 0;
  for (int g = wave; g < nblk; g += 4) {
    int nt0 = g << 1;
    v8f acc00 = {}, acc01 = {}, acc10 = {}, acc11 = {};
    for (int kt = 0; kt < ksteps; ++kt) {
      v16h a0 = frag_a(in + kt * 32, ldk);
      v16h a1 = frag_a(in + 16 * ldk + kt * 32, ldk);
      v16h b0 = frag_b(w + (nt0 * 16) * ldk + kt * 32, ldk);
      v16h b1 = frag_b(w + (nt0 * 16 + 16) * ldk + kt * 32, ldk);
      acc00 = WMMA_F16(a0, b0, acc00);   // 4 independent accumulator chains:
      acc10 = WMMA_F16(a1, b0, acc10);   // back-to-back XDL issue, loads of the
      acc01 = WMMA_F16(a0, b1, acc01);   // next K-step overlap the hazards
      acc11 = WMMA_F16(a1, b1, acc11);
    }
#pragma unroll
    for (int r = 0; r < 8; ++r) {
      out[(mrow + r) * nout + nt0 * 16 + ncol]            = (_Float16)lrelu(acc00[r]);
      out[(16 + mrow + r) * nout + nt0 * 16 + ncol]       = (_Float16)lrelu(acc10[r]);
      out[(mrow + r) * nout + (nt0 + 1) * 16 + ncol]      = (_Float16)lrelu(acc01[r]);
      out[(16 + mrow + r) * nout + (nt0 + 1) * 16 + ncol] = (_Float16)lrelu(acc11[r]);
    }
  }
}

__global__ void __launch_bounds__(128)
pointconv_mlp_kernel(const float* xyz1, const float* xyz2, const float* points2,
                     const int* knn_in,
                     const _Float16* w0h, const _Float16* w1h, const _Float16* w2h,
                     const float* wn_w0, const float* wn_b0,
                     const float* wn_w1, const float* wn_b1,
                     const float* wn_w2, const float* wn_b2,
                     _Float16* flat) {
  __shared__ __attribute__((aligned(16))) _Float16 feat[32 * C0PAD];
  __shared__ __attribute__((aligned(16))) _Float16 h0[32 * C1];
  __shared__ __attribute__((aligned(16))) _Float16 h1[32 * C1];
  __shared__ __attribute__((aligned(16))) _Float16 h2[32 * C2];
  __shared__ __attribute__((aligned(16))) _Float16 wnT[16 * 32];  // [o][nb], rows 8..15 zero
  __shared__ float dirs[32 * 3];
  __shared__ int   sidx[32];

  int p  = blockIdx.x;
  int b  = p >> 9;
  int n1 = p & (N1DIM - 1);
  int tid  = threadIdx.x;
  int wave = tid >> 5, lane = tid & 31;

  if (tid < 32) sidx[tid] = knn_in[(size_t)p * KNN + tid];
  __syncthreads();

  // gather grouped points (columns 0..127 of feat)
  for (int i = tid; i < 32 * CDIM; i += 128) {
    int nb = i >> 7, c = i & 127;
    feat[nb * C0PAD + c] =
        (_Float16)points2[(size_t)b * CDIM * N2DIM + (size_t)c * N2DIM + sidx[nb]];
  }
  // direction (cols 128..130) + zero pad (cols 131..159)
  for (int i = tid; i < 32 * 32; i += 128) {
    int nb = i >> 5, c = i & 31;
    float v = 0.f;
    if (c < 3) {
      v = xyz2[(size_t)b * 3 * N2DIM + (size_t)c * N2DIM + sidx[nb]]
        - xyz1[(size_t)b * 3 * N1DIM + (size_t)c * N1DIM + n1];
      dirs[nb * 3 + c] = v;
    }
    feat[nb * C0PAD + 128 + c] = (_Float16)v;
  }
  __syncthreads();

  // WeightNet (3 -> 8 -> 8 -> 8, ReLU) on wave 0; result transposed [o][nb]
  if (tid < 32) {
    int nb = tid;
    float t0[8], t1[8];
#pragma unroll
    for (int o = 0; o < 8; ++o) {
      float s = wn_b0[o];
#pragma unroll
      for (int c = 0; c < 3; ++c) s += wn_w0[o * 3 + c] * dirs[nb * 3 + c];
      t0[o] = fmaxf(s, 0.f);
    }
#pragma unroll
    for (int o = 0; o < 8; ++o) {
      float s = wn_b1[o];
#pragma unroll
      for (int c = 0; c < 8; ++c) s += wn_w1[o * 8 + c] * t0[c];
      t1[o] = fmaxf(s, 0.f);
    }
#pragma unroll
    for (int o = 0; o < 8; ++o) {
      float s = wn_b2[o];
#pragma unroll
      for (int c = 0; c < 8; ++c) s += wn_w2[o * 8 + c] * t1[c];
      wnT[o * 32 + nb] = (_Float16)fmaxf(s, 0.f);
    }
#pragma unroll
    for (int o = 8; o < 16; ++o) wnT[o * 32 + nb] = (_Float16)0.f;
  }

  // MLP stack (barriers also make wnT visible before agg)
  mlp_layer(feat, C0PAD, 5, w0h, h0, C1, wave, lane);
  __syncthreads();
  mlp_layer(h0, C1, 4, w1h, h1, C1, wave, lane);
  __syncthreads();
  mlp_layer(h1, C1, 4, w2h, h2, C2, wave, lane);
  __syncthreads();

  // agg[256 x 8] = h2^T (256x32) * wn (32x8, padded to 16 cols); K = 32 = one WMMA
  v16h bw = frag_b(wnT, 32);
  int ow = lane & 15;
  int mb = (lane & 16) ? 8 : 0;
  for (int mt = wave * 4; mt < wave * 4 + 4; ++mt) {
    v16h a = frag_a_t(h2 + mt * 16, C2);
    v8f acc = {};
    acc = WMMA_F16(a, bw, acc);
    if (ow < 8) {
#pragma unroll
      for (int r = 0; r < 8; ++r) {
        int ch = mt * 16 + mb + r;
        flat[(size_t)p * FDIM + ch * 8 + ow] = (_Float16)acc[r];
      }
    }
  }
}

// ---------------------------------------------------------------------------
// Kernel 3: out[B,256,N1] = lrelu( w_lin[256x2048] @ flat^T[2048 x 8192] )
// 2x2 register-blocked: each wave owns a 32x32 output block (4 WMMA tiles),
// 64 K-steps; 8 waves per 256-thread block -> 256 blocks.
// ---------------------------------------------------------------------------
__global__ void __launch_bounds__(256)
final_gemm_kernel(const _Float16* wlh, const _Float16* flat, float* out) {
  int wave = threadIdx.x >> 5, lane = threadIdx.x & 31;
  int G  = blockIdx.x * 8 + wave;     // 0 .. 2047 wave jobs
  int gm = G & 7;                     // 32-row output-channel group (256/32)
  int gn = G >> 3;                    // 32-wide point group (8192/32)
  const _Float16* a0b = wlh  + (size_t)(gm * 32) * FDIM;
  const _Float16* a1b = a0b  + (size_t)16 * FDIM;
  const _Float16* b0b = flat + (size_t)(gn * 32) * FDIM;
  const _Float16* b1b = b0b  + (size_t)16 * FDIM;

  v8f acc00 = {}, acc01 = {}, acc10 = {}, acc11 = {};
  for (int kt = 0; kt < FDIM / 32; ++kt) {
    if (kt + 8 < FDIM / 32) {         // prefetch 8 K-steps ahead (L2-hit hiding)
      __builtin_prefetch(a0b + (kt + 8) * 32, 0, 0);
      __builtin_prefetch(b0b + (kt + 8) * 32, 0, 0);
    }
    v16h a0 = frag_a(a0b + kt * 32, FDIM);
    v16h a1 = frag_a(a1b + kt * 32, FDIM);
    v16h b0 = frag_b(b0b + kt * 32, FDIM);
    v16h b1 = frag_b(b1b + kt * 32, FDIM);
    acc00 = WMMA_F16(a0, b0, acc00);
    acc10 = WMMA_F16(a1, b0, acc10);
    acc01 = WMMA_F16(a0, b1, acc01);
    acc11 = WMMA_F16(a1, b1, acc11);
  }

  int pc   = gn * 32 + (lane & 15);   // point column (tile-local col)
  int orow = gm * 32 + ((lane & 16) ? 8 : 0);
#pragma unroll
  for (int r = 0; r < 8; ++r) {
    int p0 = pc,      p1 = pc + 16;
    int o0 = orow + r, o1 = orow + 16 + r;
    out[(size_t)(p0 >> 9) * C2 * N1DIM + (size_t)o0 * N1DIM + (p0 & (N1DIM - 1))] = lrelu(acc00[r]);
    out[(size_t)(p0 >> 9) * C2 * N1DIM + (size_t)o1 * N1DIM + (p0 & (N1DIM - 1))] = lrelu(acc10[r]);
    out[(size_t)(p1 >> 9) * C2 * N1DIM + (size_t)o0 * N1DIM + (p1 & (N1DIM - 1))] = lrelu(acc01[r]);
    out[(size_t)(p1 >> 9) * C2 * N1DIM + (size_t)o1 * N1DIM + (p1 & (N1DIM - 1))] = lrelu(acc11[r]);
  }
}

// ---------------------------------------------------------------------------
extern "C" void kernel_launch(void* const* d_in, const int* in_sizes, int n_in,
                              void* d_out, int out_size, void* d_ws, size_t ws_size,
                              hipStream_t stream) {
  const float* xyz1    = (const float*)d_in[0];
  const float* xyz2    = (const float*)d_in[1];
  const float* points2 = (const float*)d_in[2];
  const float* w_mlp0  = (const float*)d_in[3];
  const float* w_mlp1  = (const float*)d_in[4];
  const float* w_mlp2  = (const float*)d_in[5];
  const float* wn_w0   = (const float*)d_in[6];
  const float* wn_b0   = (const float*)d_in[7];
  const float* wn_w1   = (const float*)d_in[8];
  const float* wn_b1   = (const float*)d_in[9];
  const float* wn_w2   = (const float*)d_in[10];
  const float* wn_b2   = (const float*)d_in[11];
  const float* w_lin   = (const float*)d_in[12];

  const int P = BDIM * N1DIM;  // 8192 query points

  char* ws = (char*)d_ws;
  size_t off = 0;
  auto take = [&](size_t bytes) -> void* {
    void* r = ws + off;
    off = (off + bytes + 255) & ~(size_t)255;
    return r;
  };
  int*      d_knn = (int*)     take((size_t)P * KNN * sizeof(int));
  _Float16* w0h   = (_Float16*)take((size_t)C1 * C0PAD * sizeof(_Float16));
  _Float16* w1h   = (_Float16*)take((size_t)C1 * C1 * sizeof(_Float16));
  _Float16* w2h   = (_Float16*)take((size_t)C2 * C1 * sizeof(_Float16));
  _Float16* wlh   = (_Float16*)take((size_t)C2 * FDIM * sizeof(_Float16));
  _Float16* flat  = (_Float16*)take((size_t)P * FDIM * sizeof(_Float16));

  const int cvt_total = C1 * C0PAD + C1 * C1 + C2 * C1 + C2 * FDIM;
  cvt_weights_kernel<<<(cvt_total + 255) / 256, 256, 0, stream>>>(
      w_mlp0, w_mlp1, w_mlp2, w_lin, w0h, w1h, w2h, wlh);

  knn_kernel<<<P, 256, 0, stream>>>(xyz1, xyz2, d_knn);

  pointconv_mlp_kernel<<<P, 128, 0, stream>>>(
      xyz1, xyz2, points2, d_knn, w0h, w1h, w2h,
      wn_w0, wn_b0, wn_w1, wn_b1, wn_w2, wn_b2, flat);

  final_gemm_kernel<<<P / 8 / 4, 256, 0, stream>>>(wlh, flat, (float*)d_out);
}
